// Gtu_35433480192031
// MI455X (gfx1250) — matrix operand order repaired and checked by reading, not compiled, not run on hardware
//
#include <hip/hip_runtime.h>
#include <math.h>

typedef unsigned short u16;
typedef unsigned int   u32;
typedef unsigned long long u64;
typedef __attribute__((ext_vector_type(16))) __bf16 v16bf;
typedef __attribute__((ext_vector_type(8)))  float  v8f;
typedef __attribute__((ext_vector_type(4)))  u32    u32x4;

struct Frag32 { u32x4 lo, hi; };   // 32 bytes == one v16bf operand

__device__ __forceinline__ u16 f2bf(float f) {
  union { float f; u32 u; } x; x.f = f;
  u32 r = x.u + 0x7FFFu + ((x.u >> 16) & 1u);   // round-to-nearest-even
  return (u16)(r >> 16);
}

__device__ __forceinline__ void wait_asynccnt0() {
#if __has_builtin(__builtin_amdgcn_s_wait_asynccnt)
  __builtin_amdgcn_s_wait_asynccnt(0);
#else
  asm volatile("s_wait_asynccnt 0x0" ::: "memory");
#endif
}

// ---------------------------------------------------------------- elementwise
__global__ void k_cvt_bf16(const float* __restrict__ in, u16* __restrict__ out, int n) {
  int i = blockIdx.x * blockDim.x + threadIdx.x;
  if (i < n) out[i] = f2bf(in[i]);
}

__global__ void k_gate_bf16(const float* __restrict__ u, const float* __restrict__ y,
                            u16* __restrict__ out, int n) {
  int i = blockIdx.x * blockDim.x + threadIdx.x;
  if (i < n) out[i] = f2bf(u[i] * y[i]);
}

// h0[p][r] = t(p) * pos_w[r] + pos_b[r];  t = [0, 1..2047, 0, -2047..-1]
__global__ void k_rpe_init(const float* __restrict__ pos_w, const float* __restrict__ pos_b,
                           float* __restrict__ h0) {
  int idx = blockIdx.x * blockDim.x + threadIdx.x;
  if (idx >= 4096 * 512) return;
  int p = idx >> 9, r = idx & 511;
  float t = (p < 2048) ? (float)p : ((p == 2048) ? 0.0f : (float)(p - 4096));
  h0[idx] = t * pos_w[r] + pos_b[r];
}

// ------------------------------------------------- LayerNorm + ReLU -> bf16
__global__ void __launch_bounds__(128)
k_ln_relu(const float* __restrict__ in, const float* __restrict__ g,
          const float* __restrict__ b, u16* __restrict__ out) {
  __shared__ float red[128];
  const int t = threadIdx.x;
  const float* x = in + (size_t)blockIdx.x * 512;
  float v[4], s = 0.f;
#pragma unroll
  for (int i = 0; i < 4; ++i) { v[i] = x[t + 128 * i]; s += v[i]; }
  red[t] = s; __syncthreads();
  for (int o = 64; o > 0; o >>= 1) { if (t < o) red[t] += red[t + o]; __syncthreads(); }
  const float mu = red[0] * (1.0f / 512.0f);
  __syncthreads();
  float sq = 0.f;
#pragma unroll
  for (int i = 0; i < 4; ++i) { float d = v[i] - mu; sq += d * d; }
  red[t] = sq; __syncthreads();
  for (int o = 64; o > 0; o >>= 1) { if (t < o) red[t] += red[t + o]; __syncthreads(); }
  const float rstd = rsqrtf(red[0] * (1.0f / 512.0f) + 1e-5f);
  u16* orow = out + (size_t)blockIdx.x * 512;
#pragma unroll
  for (int i = 0; i < 4; ++i) {
    int c = t + 128 * i;
    float y = (v[i] - mu) * rstd * g[c] + b[c];
    y = y > 0.f ? y : 0.f;
    orow[c] = f2bf(y);
  }
}

// ------------------------------------------------------------- WMMA GEMM
// C[M,Ncols] = act(A[M,K](bf16) @ W[Ncols,K](bf16)^T + bias), f32 out.
// One wave per 32x64 tile: 2 A frags x 4 B frags -> 8 WMMAs per k-step.
// Branch-free software pipeline: fragment sets s0/s1 ping-pong, loads for the
// next 64-K slab issue right after the set is consumed. Requires K % 64 == 0.
__global__ void __launch_bounds__(256)
k_gemm_bf16(const u16* __restrict__ A, const u16* __restrict__ W,
            const float* __restrict__ bias, float* __restrict__ C,
            int M, int Ncols, int K, int act) {
  const int lane = threadIdx.x & 31;
  const int wave = threadIdx.x >> 5;
  const int tiles_n = Ncols >> 6;
  const int tiles_m = M >> 5;
  const int tile = blockIdx.x * 8 + wave;
  if (tile >= tiles_m * tiles_n) return;
  const int tm = tile / tiles_n;
  const int tn = tile - tm * tiles_n;
  const int l16 = lane & 15;
  const int hi  = lane >> 4;
  const int koA = hi * 8;    // A: 16-bit layout, chunks at +0..7 and +16..23
  const int koB = hi * 8;    // B: 16-bit layout, chunks at +0..7 and +16..23
  const u16* arow0 = A + (size_t)(tm * 32 + l16) * (size_t)K;
  const u16* arow1 = arow0 + (size_t)16 * (size_t)K;
  const u16* brow[4];
#pragma unroll
  for (int j = 0; j < 4; ++j) brow[j] = W + (size_t)(tn * 64 + j * 16 + l16) * (size_t)K;

  v8f acc[8] = {};
  Frag32 fa0[2], fb0[4], fa1[2], fb1[4];

  auto LD = [&](int k, Frag32* fa_, Frag32* fb_) {
    fa_[0].lo = *(const u32x4*)(arow0 + k + koA);
    fa_[0].hi = *(const u32x4*)(arow0 + k + koA + 16);
    fa_[1].lo = *(const u32x4*)(arow1 + k + koA);
    fa_[1].hi = *(const u32x4*)(arow1 + k + koA + 16);
#pragma unroll
    for (int j = 0; j < 4; ++j) {
      fb_[j].lo = *(const u32x4*)(brow[j] + k + koB);
      fb_[j].hi = *(const u32x4*)(brow[j] + k + koB + 16);
    }
  };
  auto MMA = [&](const Frag32* fa_, const Frag32* fb_) {
    const v16bf a0 = __builtin_bit_cast(v16bf, fa_[0]);
    const v16bf a1 = __builtin_bit_cast(v16bf, fa_[1]);
#pragma unroll
    for (int j = 0; j < 4; ++j) {
      const v16bf bv = __builtin_bit_cast(v16bf, fb_[j]);
      acc[j]     = __builtin_amdgcn_wmma_f32_16x16x32_bf16(
          false, a0, false, bv, (short)0, acc[j],     false, false);
      acc[4 + j] = __builtin_amdgcn_wmma_f32_16x16x32_bf16(
          false, a1, false, bv, (short)0, acc[4 + j], false, false);
    }
  };

  // prologue: both sets in flight
  LD(0,  fa0, fb0);
  LD(32, fa1, fb1);
  // steady state: no branches, loads chase the set just consumed
  for (int k = 64; k < K; k += 64) {
    MMA(fa0, fb0);
    LD(k, fa0, fb0);
    MMA(fa1, fb1);
    LD(k + 32, fa1, fb1);
  }
  // drain
  MMA(fa0, fb0);
  MMA(fa1, fb1);

#pragma unroll
  for (int h = 0; h < 2; ++h) {
    const int rowbase = tm * 32 + h * 16 + hi * 8;
#pragma unroll
    for (int j = 0; j < 4; ++j) {
      const int col = tn * 64 + j * 16 + l16;
      const float bb = bias[col];
#pragma unroll
      for (int e = 0; e < 8; ++e) {
        float val = acc[h * 4 + j][e] + bb;
        if (act) val = val / (1.0f + __expf(-val));   // SiLU
        C[(size_t)(rowbase + e) * (size_t)Ncols + col] = val;
      }
    }
  }
}

// ------------------------------------------------------------- transpose
// per-batch: in[b][r][c] (R x C) -> out[b][c][r]
__global__ void k_transpose(const float* __restrict__ in, float* __restrict__ out,
                            int R, int C) {
  __shared__ float tile[32][33];
  const size_t base = (size_t)blockIdx.z * (size_t)R * (size_t)C;
  const float* src = in + base;
  float* dst = out + base;
  const int c0 = blockIdx.x * 32, r0 = blockIdx.y * 32;
  const int tx = threadIdx.x, ty = threadIdx.y;   // 32 x 8
#pragma unroll
  for (int i = 0; i < 32; i += 8) {
    int r = r0 + ty + i, c = c0 + tx;
    if (r < R && c < C) tile[ty + i][tx] = src[(size_t)r * C + c];
  }
  __syncthreads();
#pragma unroll
  for (int i = 0; i < 32; i += 8) {
    int r = r0 + tx, c = c0 + ty + i;
    if (r < R && c < C) dst[(size_t)c * R + r] = tile[tx][ty + i];
  }
}

// ------------------------------------------------------------- FFT (LDS)
__device__ __forceinline__ float2 cmulf(float2 a, float2 b) {
  return make_float2(a.x * b.x - a.y * b.y, a.x * b.y + a.y * b.x);
}

// Stockham radix-2, length 4096, ping-pong between s and d; result lands in s.
// sgn = -1 forward, +1 inverse (unscaled).
__device__ void fft4096(float2* __restrict__ s, float2* __restrict__ d, float sgn) {
  float2* src = s; float2* dst = d;
  int l = 2048, m = 1;
  for (int st = 0; st < 12; ++st) {
    for (int t = threadIdx.x; t < 2048; t += blockDim.x) {
      int j = t >> st;
      int k = t & (m - 1);
      float ang = sgn * 3.14159265358979323846f * (float)j / (float)l;
      float sw, cw;
      __sincosf(ang, &sw, &cw);
      float2 c0 = src[k + j * m];
      float2 c1 = src[k + j * m + 2048];
      float2 sum = make_float2(c0.x + c1.x, c0.y + c1.y);
      float2 dif = make_float2(c0.x - c1.x, c0.y - c1.y);
      dst[k + 2 * j * m]     = sum;
      dst[k + 2 * j * m + m] = cmulf(make_float2(cw, sw), dif);
    }
    __syncthreads();
    float2* tp = src; src = dst; dst = tp;
    l >>= 1; m <<= 1;
  }
}

// spectrum of kernel a: one block per channel c (1536 total)
__global__ void __launch_bounds__(256)
k_fft_a(const float* __restrict__ at, float2* __restrict__ ahat) {
  extern __shared__ float2 sm[];
  float2* Ab = sm; float2* Bb = sm + 4096;
  const float* row = at + (size_t)blockIdx.x * 4096;
  for (int i = threadIdx.x; i < 4096; i += blockDim.x)
    Ab[i] = make_float2(row[i], 0.0f);
  __syncthreads();
  fft4096(Ab, Bb, -1.0f);
  float2* orow = ahat + (size_t)blockIdx.x * 4096;
  for (int i = threadIdx.x; i < 4096; i += blockDim.x) orow[i] = Ab[i];
}

// circular conv, one block per (b, channel).
// The 32KB spectrum row is async-copied global->LDS (ASYNCcnt) while the
// forward FFT runs, then consumed after s_wait_asynccnt 0.
__global__ void __launch_bounds__(256)
k_conv(const float* __restrict__ vt, const float2* __restrict__ ahat,
       float* __restrict__ yt, int Cch) {
  extern __shared__ float2 sm[];
  float2* Ab = sm;            // FFT ping
  float2* Bb = sm + 4096;     // FFT pong
  float2* Sb = sm + 8192;     // spectrum staging (async target)
  const int wg = blockIdx.x;
  const int c = wg % Cch;
  const float* vrow = vt + (size_t)wg * 2048;
  for (int i = threadIdx.x; i < 2048; i += blockDim.x) {
    Ab[i]        = make_float2(vrow[i], 0.0f);
    Ab[i + 2048] = make_float2(0.0f, 0.0f);
  }
  // kick off async copy of ahat row (32KB) into LDS; overlaps with forward FFT
  {
    const char* gsrc = (const char*)(ahat + (size_t)c * 4096);
    const u32 lbase = (u32)(uintptr_t)Sb;
    for (int ofs = threadIdx.x * 16; ofs < 32768; ofs += 256 * 16) {
      u32 laddr = lbase + (u32)ofs;
      u64 gaddr = (u64)(gsrc + ofs);
      asm volatile("global_load_async_to_lds_b128 %0, %1, off"
                   :: "v"(laddr), "v"(gaddr) : "memory");
    }
  }
  __syncthreads();
  fft4096(Ab, Bb, -1.0f);
  wait_asynccnt0();
  __syncthreads();
  for (int i = threadIdx.x; i < 4096; i += blockDim.x)
    Ab[i] = cmulf(Ab[i], Sb[i]);
  __syncthreads();
  fft4096(Ab, Bb, 1.0f);
  float* yrow = yt + (size_t)wg * 2048;
  for (int i = threadIdx.x; i < 2048; i += blockDim.x)
    yrow[i] = Ab[i].x * (1.0f / 4096.0f);
}

// ---------------------------------------------------------------- launcher
extern "C" void kernel_launch(void* const* d_in, const int* in_sizes, int n_in,
                              void* d_out, int out_size, void* d_ws, size_t ws_size,
                              hipStream_t stream) {
  const float* x        = (const float*)d_in[0];
  const float* wu       = (const float*)d_in[1];
  const float* bu       = (const float*)d_in[2];
  const float* wv       = (const float*)d_in[3];
  const float* bv       = (const float*)d_in[4];
  const float* wo       = (const float*)d_in[5];
  const float* bo       = (const float*)d_in[6];
  const float* pos_w    = (const float*)d_in[7];
  const float* pos_b    = (const float*)d_in[8];
  const float* ln_g     = (const float*)d_in[9];
  const float* ln_b     = (const float*)d_in[10];
  const float* lw       = (const float*)d_in[11];
  const float* lb       = (const float*)d_in[12];
  const float* out_g    = (const float*)d_in[13];
  const float* out_b    = (const float*)d_in[14];
  const float* out_w    = (const float*)d_in[15];
  const float* out_bias = (const float*)d_in[16];

  const int BN = 8192, D1 = 1536, DM = 512, M2 = 4096, NCH = 1536, NB = 4, NSEQ = 2048;

  char* ws = (char*)d_ws;
  size_t off = 0;
  auto alloc = [&](size_t bytes) { size_t r = off; off += (bytes + 255) & ~(size_t)255; return r; };

  const size_t o_xbf  = alloc((size_t)BN * DM * 2);
  const size_t o_wubf = alloc((size_t)D1 * DM * 2);
  const size_t o_wvbf = alloc((size_t)D1 * DM * 2);
  const size_t o_wobf = alloc((size_t)DM * D1 * 2);
  const size_t o_lwbf = alloc((size_t)3 * 512 * 512 * 2);
  const size_t o_owbf = alloc((size_t)D1 * DM * 2);
  const size_t o_u    = alloc((size_t)BN * D1 * 4);
  const size_t o_v    = alloc((size_t)BN * D1 * 4);      // later reused as y_t
  const size_t o_vt   = alloc((size_t)BN * D1 * 4);
  const size_t o_h0   = alloc((size_t)M2 * 512 * 4);
  const size_t o_h1   = alloc((size_t)M2 * 512 * 4);
  const size_t o_hbf  = alloc((size_t)M2 * 512 * 2);
  const size_t o_a    = alloc((size_t)M2 * D1 * 4);      // a, then first half of y
  const size_t o_at   = alloc((size_t)M2 * D1 * 4);      // a^T, then second half of y
  const size_t o_ahat = alloc((size_t)NCH * 4096 * 8);
  const size_t o_gbf  = alloc((size_t)BN * D1 * 2);
  (void)ws_size; (void)in_sizes; (void)n_in; (void)out_size;

  auto cvt = [&](const float* src, size_t dsto, int n) {
    k_cvt_bf16<<<(n + 255) / 256, 256, 0, stream>>>(src, (u16*)(ws + dsto), n);
  };
  auto gemm = [&](size_t Ao, size_t Wo, const float* bias, float* C,
                  int M, int Nc, int K, int act) {
    int waves = (M / 32) * (Nc / 64);
    k_gemm_bf16<<<(waves + 7) / 8, 256, 0, stream>>>(
        (const u16*)(ws + Ao), (const u16*)(ws + Wo), bias, C, M, Nc, K, act);
  };

  // 1) bf16 conversions (weights + x)
  cvt(x,     o_xbf,  BN * DM);
  cvt(wu,    o_wubf, D1 * DM);
  cvt(wv,    o_wvbf, D1 * DM);
  cvt(wo,    o_wobf, DM * D1);
  cvt(lw,    o_lwbf, 3 * 512 * 512);
  cvt(out_w, o_owbf, D1 * DM);

  // 2) u = silu(x@wu^T+bu), v = silu(x@wv^T+bv)
  gemm(o_xbf, o_wubf, bu, (float*)(ws + o_u), BN, D1, DM, 1);
  gemm(o_xbf, o_wvbf, bv, (float*)(ws + o_v), BN, D1, DM, 1);

  // 3) v (B,N,D1) -> v_t (B,D1,N)
  {
    dim3 g(D1 / 32, NSEQ / 32, NB), blk(32, 8);
    k_transpose<<<g, blk, 0, stream>>>((const float*)(ws + o_v), (float*)(ws + o_vt), NSEQ, D1);
  }

  // 4) RPE MLP -> kernel a (4096, 1536)
  k_rpe_init<<<(M2 * 512 + 255) / 256, 256, 0, stream>>>(pos_w, pos_b, (float*)(ws + o_h0));
  size_t hcur = o_h0, hnext = o_h1;
  for (int i = 0; i < 3; ++i) {
    k_ln_relu<<<M2, 128, 0, stream>>>((const float*)(ws + hcur), ln_g + i * 512,
                                      ln_b + i * 512, (u16*)(ws + o_hbf));
    gemm(o_hbf, o_lwbf + (size_t)i * 512 * 512 * 2, lb + i * 512,
         (float*)(ws + hnext), M2, 512, 512, 0);
    size_t t = hcur; hcur = hnext; hnext = t;
  }
  k_ln_relu<<<M2, 128, 0, stream>>>((const float*)(ws + hcur), out_g, out_b, (u16*)(ws + o_hbf));
  gemm(o_hbf, o_owbf, out_bias, (float*)(ws + o_a), M2, D1, 512, 0);

  // 5) a (4096,1536) -> a_t (1536,4096); spectra ahat
  {
    dim3 g(D1 / 32, M2 / 32, 1), blk(32, 8);
    k_transpose<<<g, blk, 0, stream>>>((const float*)(ws + o_a), (float*)(ws + o_at), M2, D1);
  }
  k_fft_a<<<NCH, 256, 65536, stream>>>((const float*)(ws + o_at), (float2*)(ws + o_ahat));

  // 6) per-(b,channel) circular convolution; y_t written over old v buffer
  k_conv<<<NB * NCH, 256, 98304, stream>>>((const float*)(ws + o_vt),
                                           (const float2*)(ws + o_ahat),
                                           (float*)(ws + o_v), NCH);

  // 7) y_t (B,D1,N) -> y (B,N,D1), written over (a, a_t) region
  {
    dim3 g(NSEQ / 32, D1 / 32, NB), blk(32, 8);
    k_transpose<<<g, blk, 0, stream>>>((const float*)(ws + o_v), (float*)(ws + o_a), D1, NSEQ);
  }

  // 8) gate: g = bf16(u * y)
  k_gate_bf16<<<(BN * D1 + 255) / 256, 256, 0, stream>>>(
      (const float*)(ws + o_u), (const float*)(ws + o_a), (u16*)(ws + o_gbf), BN * D1);

  // 9) out = g @ wo^T + bo
  gemm(o_gbf, o_wobf, bo, (float*)d_out, BN, DM, D1, 0);
}
